// TransformerBlock_38886633898768
// MI455X (gfx1250) — matrix operand order repaired and checked by reading, not compiled, not run on hardware
//
#include <hip/hip_runtime.h>

// ---------------------------------------------------------------------------
// Transformer block forward on MI455X (gfx1250, wave32, WMMA).
// All matmuls in bf16 via v_wmma_f32_16x16x32_bf16 (f32 accumulate).
// GEMM global->LDS staging uses CDNA5 async-to-LDS DMA (ASYNCcnt) with a
// double-buffered software pipeline.
// ---------------------------------------------------------------------------

typedef __bf16 bf16;
typedef __attribute__((ext_vector_type(16))) __bf16 v16bf;
typedef __attribute__((ext_vector_type(8)))  float  v8f;

// Problem constants (match reference)
static constexpr int Bb   = 4;
static constexpr int Tt   = 2048;
static constexpr int Cc   = 1024;
static constexpr int Hh   = 16;
static constexpr int HS   = 64;
static constexpr int DFF  = 4096;
static constexpr int NTOK = Bb * Tt; // 8192

// ---------------------------------------------------------------------------
// CDNA5 async global->LDS copy (per-lane 16B), tracked by ASYNCcnt.
// lds byte address = low 32 bits of the flat shared pointer (LDS aperture
// stores the offset in ADDR[31:0]).
// ---------------------------------------------------------------------------
__device__ __forceinline__ uint32_t lds_off(const void* p) {
  return (uint32_t)(uintptr_t)p;
}

__device__ __forceinline__ void async_copy_b128(uint32_t lds_addr,
                                                const bf16* gaddr) {
  asm volatile("global_load_async_to_lds_b128 %0, %1, off"
               :
               : "v"(lds_addr), "v"(gaddr)
               : "memory");
}

__device__ __forceinline__ void wait_async0() {
  asm volatile("s_wait_asynccnt 0" ::: "memory");
}

// ---------------------------------------------------------------------------
// WMMA fragment helpers (wave32; A and B both use k = 16*(lane/16) + element)
// ---------------------------------------------------------------------------

__device__ __forceinline__ v8f wmma_bf16(v16bf a, v16bf b, v8f c) {
  return __builtin_amdgcn_wmma_f32_16x16x32_bf16(
      /*neg_a=*/false, a, /*neg_b=*/false, b,
      /*c_mod=*/(short)0, c, /*reuse_a=*/false, /*reuse_b=*/false);
}

// A-matrix 16x32 (MxK), row-major source with leading dim lda (elements).
__device__ __forceinline__ v16bf load_a_frag(const bf16* __restrict__ base, int lda) {
  const int lane = threadIdx.x & 31;
  const int m  = lane & 15;
  const int kb = (lane >> 4) << 4;
  const bf16* p0 = base + (size_t)m * lda + kb;
  const bf16* p1 = p0 + 8;
  v16bf a;
#pragma unroll
  for (int i = 0; i < 8; ++i) a[i] = p0[i];
#pragma unroll
  for (int i = 0; i < 8; ++i) a[8 + i] = p1[i];
  return a;
}

// B-matrix 32x16 (KxN) supplied N-major (Bt[n][k], leading dim ldk).
__device__ __forceinline__ v16bf load_b_fragT(const bf16* __restrict__ base, int ldk) {
  const int lane = threadIdx.x & 31;
  const int n  = lane & 15;
  const int kb = (lane >> 4) << 4;
  const bf16* p = base + (size_t)n * ldk + kb;
  v16bf b;
#pragma unroll
  for (int i = 0; i < 16; ++i) b[i] = p[i];
  return b;
}

// ---------------------------------------------------------------------------
// Weight transpose + fp32 -> bf16:  W[K][N] -> WT[N][K]
// ---------------------------------------------------------------------------
__global__ __launch_bounds__(256) void transpose_f32_to_bf16(
    const float* __restrict__ W, bf16* __restrict__ WT, int K, int N) {
  __shared__ float tile[32][33];
  const int n0 = blockIdx.x * 32, k0 = blockIdx.y * 32;
#pragma unroll
  for (int i = threadIdx.y; i < 32; i += 8)
    tile[i][threadIdx.x] = W[(size_t)(k0 + i) * N + n0 + threadIdx.x];
  __syncthreads();
#pragma unroll
  for (int i = threadIdx.y; i < 32; i += 8)
    WT[(size_t)(n0 + i) * K + k0 + threadIdx.x] = (bf16)tile[threadIdx.x][i];
}

// ---------------------------------------------------------------------------
// LayerNorm (fp32 in -> bf16 out), one 256-thread block per token row (C=1024)
// ---------------------------------------------------------------------------
__global__ __launch_bounds__(256) void layernorm_bf16(
    const float* __restrict__ x, const float* __restrict__ g,
    const float* __restrict__ be, bf16* __restrict__ out) {
  const int row = blockIdx.x;
  const float* xr = x + (size_t)row * Cc;
  float vals[4];
  float lsum = 0.f, lsq = 0.f;
#pragma unroll
  for (int i = 0; i < 4; ++i) {
    float v = xr[i * 256 + threadIdx.x];
    vals[i] = v; lsum += v; lsq += v * v;
  }
#pragma unroll
  for (int off = 16; off; off >>= 1) {
    lsum += __shfl_xor(lsum, off, 32);
    lsq  += __shfl_xor(lsq,  off, 32);
  }
  __shared__ float sred[2][8];
  const int wave = threadIdx.x >> 5, lane = threadIdx.x & 31;
  if (lane == 0) { sred[0][wave] = lsum; sred[1][wave] = lsq; }
  __syncthreads();
  float ts = 0.f, tq = 0.f;
#pragma unroll
  for (int w = 0; w < 8; ++w) { ts += sred[0][w]; tq += sred[1][w]; }
  const float mu = ts * (1.f / Cc);
  const float var = tq * (1.f / Cc) - mu * mu;
  const float rstd = rsqrtf(var + 1e-5f);
  bf16* orow = out + (size_t)row * Cc;
#pragma unroll
  for (int i = 0; i < 4; ++i) {
    int c = i * 256 + threadIdx.x;
    orow[c] = (bf16)((vals[i] - mu) * rstd * g[c] + be[c]);
  }
}

// ---------------------------------------------------------------------------
// WMMA GEMM: Out = epilogue( X[M][K](bf16) * WT[N][K](bf16) + bias )
// Tiles: BM=64, BN=128, BK=32; 8 waves, each a 32x32 sub-tile.
// Double-buffered async-to-LDS pipeline: issue tile i+1, compute tile i.
// EPI: 0 = scatter bf16 to [B,H,T,HS] (QKV)
//      1 = fp32 out = resid + gemm + bias   (O-proj / FF2)
//      2 = bf16 out = relu(gemm + bias)     (FF1)
// ---------------------------------------------------------------------------
template <int EPI>
__global__ __launch_bounds__(256) void gemm_wmma(
    const bf16*  __restrict__ X,  const bf16* __restrict__ WT,
    const float* __restrict__ bias, const float* __restrict__ resid,
    float* __restrict__ outF, bf16* __restrict__ outB,
    int M, int N, int K) {
  constexpr int BM = 64, BN = 128, BK = 32;
  __shared__ alignas(16) bf16 As[2][BM][BK];
  __shared__ alignas(16) bf16 Bs[2][BN][BK];

  const int tid  = threadIdx.x;
  const int wave = tid >> 5;
  const int wm   = wave >> 2;   // 0..1  (rows of 32)
  const int wn   = wave & 3;    // 0..3  (cols of 32)
  const int m0 = blockIdx.y * BM;
  const int n0 = blockIdx.x * BN;

  v8f acc[2][2] = {};

  const int ar = tid >> 2;          // 0..63
  const int ac = (tid & 3) << 3;    // 0,8,16,24
  const int br = tid >> 1;          // 0..127
  const int bc = (tid & 1) << 4;    // 0,16

  const bf16* aSrc = X  + (size_t)(m0 + ar) * K + ac;
  const bf16* bSrc = WT + (size_t)(n0 + br) * K + bc;
  const uint32_t aDst[2] = { lds_off(&As[0][ar][ac]), lds_off(&As[1][ar][ac]) };
  const uint32_t bDst[2] = { lds_off(&Bs[0][br][bc]), lds_off(&Bs[1][br][bc]) };

  // prologue: stage tile 0 via async DMA
  async_copy_b128(aDst[0], aSrc);
  async_copy_b128(bDst[0], bSrc);
  async_copy_b128(bDst[0] + 16, bSrc + 8);

  const int niter = K / BK;
  for (int i = 0; i < niter; ++i) {
    wait_async0();        // this wave's tile-i DMA done
    __syncthreads();      // all waves' tile-i data visible; prev reads retired
    if (i + 1 < niter) {  // issue tile i+1 into the other buffer
      const int nb = (i + 1) & 1;
      const bf16* aN = aSrc + (size_t)(i + 1) * BK;
      const bf16* bN = bSrc + (size_t)(i + 1) * BK;
      async_copy_b128(aDst[nb], aN);
      async_copy_b128(bDst[nb], bN);
      async_copy_b128(bDst[nb] + 16, bN + 8);
    }
    const int cb = i & 1;
    v16bf a0 = load_a_frag(&As[cb][wm * 32 +  0][0], BK);
    v16bf a1 = load_a_frag(&As[cb][wm * 32 + 16][0], BK);
    v16bf b0 = load_b_fragT(&Bs[cb][wn * 32 +  0][0], BK);
    v16bf b1 = load_b_fragT(&Bs[cb][wn * 32 + 16][0], BK);
    acc[0][0] = wmma_bf16(a0, b0, acc[0][0]);
    acc[0][1] = wmma_bf16(a0, b1, acc[0][1]);
    acc[1][0] = wmma_bf16(a1, b0, acc[1][0]);
    acc[1][1] = wmma_bf16(a1, b1, acc[1][1]);
  }

  // Epilogue: C layout -> lane n = lane&15, row m = v + 8*(lane>=16)
  const int lane = tid & 31;
  const int cn   = lane & 15;
  const int half = lane >> 4;
#pragma unroll
  for (int mi = 0; mi < 2; ++mi)
#pragma unroll
    for (int ni = 0; ni < 2; ++ni)
#pragma unroll
      for (int vv = 0; vv < 8; ++vv) {
        const int gm = m0 + wm * 32 + mi * 16 + vv + half * 8;
        const int gn = n0 + wn * 32 + ni * 16 + cn;
        float val = acc[mi][ni][vv] + bias[gn];
        if constexpr (EPI == 0) {
          // gm = b*T + t ; gn = h*HS + d  ->  [B,H,T,HS]
          const int bb = gm >> 11, t = gm & (Tt - 1);
          const int hh = gn >> 6,  d = gn & (HS - 1);
          outB[(((size_t)(bb * Hh + hh)) * Tt + t) * HS + d] = (bf16)val;
        } else if constexpr (EPI == 1) {
          const size_t idx = (size_t)gm * N + gn;
          outF[idx] = resid[idx] + val;
        } else {
          outB[(size_t)gm * N + gn] = (bf16)fmaxf(val, 0.f);
        }
      }
}

// ---------------------------------------------------------------------------
// Flash attention: grid (T/128, B*H); 8 waves x 16 query rows each.
// S = Q K^T via WMMA, online softmax (f32), O += P V via WMMA (P re-staged
// through per-wave LDS to convert C-layout -> A-fragment layout).
// ---------------------------------------------------------------------------
__global__ __launch_bounds__(256) void attn_flash(
    const bf16* __restrict__ Q, const bf16* __restrict__ Kk,
    const bf16* __restrict__ V, bf16* __restrict__ Y) {
  __shared__ alignas(16) bf16 Ks[64][64];      // [key][hs]  (N-major for S)
  __shared__ alignas(16) bf16 Vt[64][64];      // [hs][key]  (N-major for PV)
  __shared__ alignas(16) bf16 Ps[8][16][64];   // per-wave P tile

  const int tid  = threadIdx.x;
  const int wave = tid >> 5, lane = tid & 31;
  const int cn = lane & 15, half = lane >> 4;
  const int bh = blockIdx.y;
  const int bb = bh / Hh, hh = bh % Hh;
  const int q0  = blockIdx.x * 128;
  const int qr0 = q0 + wave * 16;

  const bf16* Qh = Q  + (size_t)bh * Tt * HS;
  const bf16* Kh = Kk + (size_t)bh * Tt * HS;
  const bf16* Vh = V  + (size_t)bh * Tt * HS;

  const v16bf aQ0 = load_a_frag(Qh + (size_t)qr0 * HS,      HS);
  const v16bf aQ1 = load_a_frag(Qh + (size_t)qr0 * HS + 32, HS);

  v8f oacc[4] = {};
  float m_i[8], l_i[8];
#pragma unroll
  for (int vv = 0; vv < 8; ++vv) { m_i[vv] = -1e30f; l_i[vv] = 0.f; }

  const int sr = tid >> 2;          // 0..63
  const int sc = (tid & 3) << 4;    // 0,16,32,48

  const int nkb = q0 / 64 + 2;      // causal key-block bound
  for (int kb = 0; kb < nkb; ++kb) {
    // stage K tile (async DMA, direct) and V tile (transposed via VGPRs)
    {
      async_copy_b128(lds_off(&Ks[sr][sc]),
                      Kh + (size_t)(kb * 64 + sr) * HS + sc);
      async_copy_b128(lds_off(&Ks[sr][sc + 8]),
                      Kh + (size_t)(kb * 64 + sr) * HS + sc + 8);
      alignas(16) bf16 tmp[16];
      const uint4* vs = (const uint4*)(Vh + (size_t)(kb * 64 + sr) * HS + sc);
      *(uint4*)&tmp[0] = vs[0];
      *(uint4*)&tmp[8] = vs[1];
#pragma unroll
      for (int i = 0; i < 16; ++i) Vt[sc + i][sr] = tmp[i];
      wait_async0();
    }
    __syncthreads();

    // S = Q K^T  (16 queries x 64 keys)
    v8f S[4];
#pragma unroll
    for (int nt = 0; nt < 4; ++nt) {
      v16bf b0 = load_b_fragT(&Ks[nt * 16][0],  64);
      v16bf b1 = load_b_fragT(&Ks[nt * 16][32], 64);
      v8f s = {};
      s = wmma_bf16(aQ0, b0, s);
      s = wmma_bf16(aQ1, b1, s);
      S[nt] = s;
    }

    // scale + causal mask + row max
    float rm[8];
#pragma unroll
    for (int vv = 0; vv < 8; ++vv) rm[vv] = -1e30f;
#pragma unroll
    for (int nt = 0; nt < 4; ++nt) {
      const int kc = kb * 64 + nt * 16 + cn;
#pragma unroll
      for (int vv = 0; vv < 8; ++vv) {
        const int qr = qr0 + vv + half * 8;
        float sv = S[nt][vv] * 0.125f;        // 1/sqrt(64)
        if (kc > qr) sv = -1e30f;
        S[nt][vv] = sv;
        rm[vv] = fmaxf(rm[vv], sv);
      }
    }
#pragma unroll
    for (int vv = 0; vv < 8; ++vv)
#pragma unroll
      for (int off = 1; off < 16; off <<= 1)
        rm[vv] = fmaxf(rm[vv], __shfl_xor(rm[vv], off, 32));

    float alpha[8], rs[8];
#pragma unroll
    for (int vv = 0; vv < 8; ++vv) {
      const float nm = fmaxf(m_i[vv], rm[vv]);
      alpha[vv] = __expf(m_i[vv] - nm);
      m_i[vv] = nm;
      rs[vv] = 0.f;
    }
#pragma unroll
    for (int nt = 0; nt < 4; ++nt)
#pragma unroll
      for (int vv = 0; vv < 8; ++vv) {
        const float p = __expf(S[nt][vv] - m_i[vv]);
        S[nt][vv] = p;
        rs[vv] += p;
      }
#pragma unroll
    for (int vv = 0; vv < 8; ++vv) {
#pragma unroll
      for (int off = 1; off < 16; off <<= 1)
        rs[vv] += __shfl_xor(rs[vv], off, 32);
      l_i[vv] = l_i[vv] * alpha[vv] + rs[vv];
    }
#pragma unroll
    for (int nt = 0; nt < 4; ++nt)
#pragma unroll
      for (int vv = 0; vv < 8; ++vv)
        oacc[nt][vv] *= alpha[vv];

    // P -> per-wave LDS (C layout -> row-major), then reload as A fragments
#pragma unroll
    for (int nt = 0; nt < 4; ++nt)
#pragma unroll
      for (int vv = 0; vv < 8; ++vv)
        Ps[wave][vv + half * 8][nt * 16 + cn] = (bf16)S[nt][vv];

    const v16bf aP0 = load_a_frag(&Ps[wave][0][0],  64);
    const v16bf aP1 = load_a_frag(&Ps[wave][0][32], 64);
#pragma unroll
    for (int nt = 0; nt < 4; ++nt) {
      v16bf bv0 = load_b_fragT(&Vt[nt * 16][0],  64);
      v16bf bv1 = load_b_fragT(&Vt[nt * 16][32], 64);
      oacc[nt] = wmma_bf16(aP0, bv0, oacc[nt]);
      oacc[nt] = wmma_bf16(aP1, bv1, oacc[nt]);
    }
    __syncthreads();
  }

  // finalize: O /= l, write merged-head layout [B,T,C] bf16
#pragma unroll
  for (int vv = 0; vv < 8; ++vv) l_i[vv] = 1.f / l_i[vv];
#pragma unroll
  for (int nt = 0; nt < 4; ++nt)
#pragma unroll
    for (int vv = 0; vv < 8; ++vv) {
      const int qr = qr0 + vv + half * 8;
      Y[((size_t)bb * Tt + qr) * Cc + hh * HS + nt * 16 + cn] =
          (bf16)(oacc[nt][vv] * l_i[vv]);
    }
}

// ---------------------------------------------------------------------------
// Launch
// ---------------------------------------------------------------------------
extern "C" void kernel_launch(void* const* d_in, const int* in_sizes, int n_in,
                              void* d_out, int out_size, void* d_ws, size_t ws_size,
                              hipStream_t stream) {
  const float* x  = (const float*)d_in[0];
  const float* Wq = (const float*)d_in[1];
  const float* bq = (const float*)d_in[2];
  const float* Wk = (const float*)d_in[3];
  const float* bk = (const float*)d_in[4];
  const float* Wv = (const float*)d_in[5];
  const float* bv = (const float*)d_in[6];
  const float* Wo = (const float*)d_in[7];
  const float* bo = (const float*)d_in[8];
  const float* W1 = (const float*)d_in[9];
  const float* b1 = (const float*)d_in[10];
  const float* W2 = (const float*)d_in[11];
  const float* b2 = (const float*)d_in[12];
  const float* g1 = (const float*)d_in[13];
  const float* be1 = (const float*)d_in[14];
  const float* g2 = (const float*)d_in[15];
  const float* be2 = (const float*)d_in[16];

  // workspace carve-out
  char* ws = (char*)d_ws;
  auto carve = [&](size_t bytes) {
    char* p = ws;
    ws += (bytes + 255) & ~(size_t)255;
    return p;
  };
  bf16* WqT = (bf16*)carve((size_t)Cc * Cc * 2);
  bf16* WkT = (bf16*)carve((size_t)Cc * Cc * 2);
  bf16* WvT = (bf16*)carve((size_t)Cc * Cc * 2);
  bf16* WoT = (bf16*)carve((size_t)Cc * Cc * 2);
  bf16* W1T = (bf16*)carve((size_t)DFF * Cc * 2);
  bf16* W2T = (bf16*)carve((size_t)Cc * DFF * 2);
  bf16* X1  = (bf16*)carve((size_t)NTOK * Cc * 2);     // LN1 out
  bf16* Qb  = (bf16*)carve((size_t)NTOK * Cc * 2);
  bf16* Kb  = (bf16*)carve((size_t)NTOK * Cc * 2);
  bf16* Vb  = (bf16*)carve((size_t)NTOK * Cc * 2);
  bf16* Yb  = (bf16*)carve((size_t)NTOK * Cc * 2);     // attn out
  float* Xr = (float*)carve((size_t)NTOK * Cc * 4);    // x + attn (fp32)
  bf16* X2  = (bf16*)carve((size_t)NTOK * Cc * 2);     // LN2 out
  bf16* Hf  = (bf16*)carve((size_t)NTOK * DFF * 2);    // relu(ff1)

  const dim3 tb32(32, 8);
  // weight transposes (fp32 -> bf16, N-major)
  transpose_f32_to_bf16<<<dim3(Cc / 32, Cc / 32), tb32, 0, stream>>>(Wq, WqT, Cc, Cc);
  transpose_f32_to_bf16<<<dim3(Cc / 32, Cc / 32), tb32, 0, stream>>>(Wk, WkT, Cc, Cc);
  transpose_f32_to_bf16<<<dim3(Cc / 32, Cc / 32), tb32, 0, stream>>>(Wv, WvT, Cc, Cc);
  transpose_f32_to_bf16<<<dim3(Cc / 32, Cc / 32), tb32, 0, stream>>>(Wo, WoT, Cc, Cc);
  transpose_f32_to_bf16<<<dim3(DFF / 32, Cc / 32), tb32, 0, stream>>>(W1, W1T, Cc, DFF);
  transpose_f32_to_bf16<<<dim3(Cc / 32, DFF / 32), tb32, 0, stream>>>(W2, W2T, DFF, Cc);

  // LN1
  layernorm_bf16<<<NTOK, 256, 0, stream>>>(x, g1, be1, X1);

  // QKV projections (scatter into [B,H,T,HS])
  const dim3 gC(Cc / 128, NTOK / 64);
  gemm_wmma<0><<<gC, 256, 0, stream>>>(X1, WqT, bq, nullptr, nullptr, Qb, NTOK, Cc, Cc);
  gemm_wmma<0><<<gC, 256, 0, stream>>>(X1, WkT, bk, nullptr, nullptr, Kb, NTOK, Cc, Cc);
  gemm_wmma<0><<<gC, 256, 0, stream>>>(X1, WvT, bv, nullptr, nullptr, Vb, NTOK, Cc, Cc);

  // causal flash attention
  attn_flash<<<dim3(Tt / 128, Bb * Hh), 256, 0, stream>>>(Qb, Kb, Vb, Yb);

  // output projection + residual (fp32)
  gemm_wmma<1><<<gC, 256, 0, stream>>>(Yb, WoT, bo, x, Xr, nullptr, NTOK, Cc, Cc);

  // LN2
  layernorm_bf16<<<NTOK, 256, 0, stream>>>(Xr, g2, be2, X2);

  // FF1 (+bias, ReLU) -> bf16
  gemm_wmma<2><<<dim3(DFF / 128, NTOK / 64), 256, 0, stream>>>(
      X2, W1T, b1, nullptr, nullptr, Hf, NTOK, DFF, Cc);

  // FF2 (+bias) + residual -> d_out (fp32)
  gemm_wmma<1><<<gC, 256, 0, stream>>>(Hf, W2T, b2, Xr, (float*)d_out, nullptr,
                                       NTOK, Cc, DFF);

  (void)in_sizes; (void)n_in; (void)out_size; (void)ws_size;
}
